// EntityEncoder_77154792506025
// MI455X (gfx1250) — matrix operand order repaired and checked by reading, not compiled, not run on hardware
//
#include <hip/hip_runtime.h>

// Problem constants (reference: B=32, P=16, E=50000, H=128)
#define B_    32
#define P_    16
#define E_    50000
#define H_    128
#define M_    (B_ * P_)          // 512 rows of the big GEMM
#define KC    28                 // K chunks (grid.y of kernel 1)
#define KCH   1792               // chunk length, multiple of 64; KC*KCH = 50176 >= E_
#define KB32  ((KC * KCH) / 32)  // 1568 32-wide K blocks
#define NR    (KCH / 64)         // 28 staging rounds per chunk (even)
#define ASTR  72                 // padded LDS row stride (halves): 144B -> conflict-minimal
#define EPS_  1e-5f

typedef __attribute__((ext_vector_type(16))) _Float16 v16h;
typedef __attribute__((ext_vector_type(8)))  _Float16 v8h;
typedef __attribute__((ext_vector_type(8)))  float    v8f;

// ---------------------------------------------------------------------------
// Workgroup barrier that orders ONLY LDS traffic (DScnt), leaving in-flight
// global loads (register prefetches) outstanding across the barrier.
// __syncthreads() would conservatively drain loadcnt too, killing the
// prefetch pipeline.
// ---------------------------------------------------------------------------
__device__ __forceinline__ void ee_barrier_ds() {
  asm volatile(
    "s_wait_dscnt 0x0\n\t"
    "s_barrier_signal -1\n\t"
    "s_barrier_wait -1"
    ::: "memory");
}

// ---------------------------------------------------------------------------
// Kernel 0: pre-swizzle emb (f32, row-major [E][H]) into f16 B-fragment order:
//   embSw[kb][hs][n][h] = emb0[kb*32 + hs*16 + h][n]
// Zeros baked in for padding_idx row 0 and the padded K tail.
// ---------------------------------------------------------------------------
__global__ __launch_bounds__(256)
void ee_embswizzle(const float* __restrict__ emb, _Float16* __restrict__ embSw) {
  const int kb = blockIdx.x;          // 0..KB32-1
  const int t  = threadIdx.x;
  const int hs = t >> 7;              // 0,1
  const int n  = t & (H_ - 1);
  v16h o;
  #pragma unroll
  for (int j = 0; j < 16; ++j) {
    int e = kb * 32 + hs * 16 + j;
    float v = 0.f;
    if (e > 0 && e < E_) v = emb[(size_t)e * H_ + n];
    o[j] = (_Float16)v;
  }
  *(v16h*)(embSw + (((size_t)kb * 2 + hs) * H_ + n) * 16) = o;
}

// ---------------------------------------------------------------------------
// Kernel 1 helpers (force-inlined; constant indices only -> no scratch).
// ---------------------------------------------------------------------------
__device__ __forceinline__
void ee_prefetch(int4 (&pre)[4], const int* __restrict__ inputs,
                 size_t rowBase, int kn, int cb) {
  #pragma unroll
  for (int i = 0; i < 4; ++i) {
    int start = kn + cb + i * 4;
    int ecl = (start > E_ - 4) ? (E_ - 4) : start;  // E_%4==0: groups never straddle
    pre[i] = *(const int4*)(inputs + rowBase + ecl);
  }
}

__device__ __forceinline__
void ee_stage(const int4 (&pre)[4], int k, int cb, int r,
              _Float16* buf, int& cntAcc) {
  v8h p0, p1;
  #pragma unroll
  for (int i = 0; i < 4; ++i) {
    bool valid = (k + cb + i * 4) < E_;             // whole-group validity
    int vv[4] = {pre[i].x, pre[i].y, pre[i].z, pre[i].w};
    #pragma unroll
    for (int j = 0; j < 4; ++j) {
      bool on = valid && (vv[j] == 1);
      _Float16 hv = on ? (_Float16)1.f : (_Float16)0.f;
      if (i < 2) p0[i * 4 + j] = hv; else p1[(i - 2) * 4 + j] = hv;
      cntAcc += on ? 1 : 0;
    }
  }
  *(v8h*)&buf[r * ASTR + cb]     = p0;
  *(v8h*)&buf[r * ASTR + cb + 8] = p1;
}

__device__ __forceinline__
void ee_compute(const _Float16* buf, const _Float16* __restrict__ embSw,
                int kbBase, int halfSel, int arow, int ncol, v8f (&acc)[4]) {
  #pragma unroll
  for (int ksub = 0; ksub < 2; ++ksub) {
    const int kb = kbBase + ksub;
    v16h bf = *(const v16h*)(embSw + (((size_t)kb * 2 + halfSel) * H_ + ncol) * 16);
    v16h af[4];
    #pragma unroll
    for (int s = 0; s < 4; ++s) {
      const _Float16* ap = &buf[(s * 16 + arow) * ASTR + ksub * 32 + halfSel * 16];
      v8h lo = *(const v8h*)ap;
      v8h hi = *(const v8h*)(ap + 8);
      af[s] = __builtin_shufflevector(lo, hi,
                0,1,2,3,4,5,6,7,8,9,10,11,12,13,14,15);
    }
    #pragma unroll
    for (int s = 0; s < 4; ++s)
      acc[s] = __builtin_amdgcn_wmma_f32_16x16x32_f16(
                 false, af[s], false, bf, (short)0, acc[s], false, false);
  }
}

// ---------------------------------------------------------------------------
// Kernel 1: partial masked GEMM via V_WMMA_F32_16X16X32_F16.
// Block = 256 threads = 8 waves; 64(M) x 128(N) tile over one K chunk.
// Wave w owns N columns [16w,16w+16); reuses each B fragment for 4 M subtiles.
// Depth-2 register prefetch issued AFTER the DS-only barrier so the loads
// stay outstanding across it; double-buffered LDS, one barrier per round.
// ---------------------------------------------------------------------------
__global__ __launch_bounds__(256)
void ee_wmma_partial(const int* __restrict__ inputs,
                     const _Float16* __restrict__ embSw,
                     float* __restrict__ ps /*[KC][M_][H_]*/,
                     float* __restrict__ pc /*[KC][M_]*/) {
  __shared__ __align__(16) _Float16 lsA[2][64 * ASTR];
  __shared__ float lsCnt[256];

  const int mt   = blockIdx.x;        // 0..7
  const int kc   = blockIdx.y;        // 0..KC-1
  const int t    = threadIdx.x;
  const int lane = t & 31;
  const int wave = t >> 5;
  const int n0   = wave * 16;
  const int m0   = mt * 64;

  // staging: thread t handles row r, 16 K-columns starting q*16
  const int r    = t >> 2;            // 0..63
  const int q    = t & 3;             // 0..3
  const int cb   = q * 16;
  const int mrow = m0 + r;
  const size_t rowBase = (size_t)mrow * E_;

  const int halfSel = lane >> 4;      // which K half this lane holds
  const int arow    = lane & 15;      // A row within subtile
  const int ncol    = n0 + arow;      // B/D column
  const int kbeg    = kc * KCH;
  const int kb0     = kbeg >> 5;      // first 32-K block index

  v8f acc[4] = {};
  int  cntAcc = 0;

  int4 preA[4], preB[4];
  ee_prefetch(preA, inputs, rowBase, kbeg,      cb);
  ee_prefetch(preB, inputs, rowBase, kbeg + 64, cb);

  for (int round = 0; round < NR; round += 2) {
    const int k = kbeg + round * 64;

    // ---- even round: buffer 0 ----
    ee_stage(preA, k, cb, r, &lsA[0][0], cntAcc);
    ee_barrier_ds();
    ee_prefetch(preA, inputs, rowBase, k + 128, cb);   // outstanding across barriers
    ee_compute(&lsA[0][0], embSw, kb0 + round * 2, halfSel, arow, ncol, acc);

    // ---- odd round: buffer 1 ----
    ee_stage(preB, k + 64, cb, r, &lsA[1][0], cntAcc);
    ee_barrier_ds();
    ee_prefetch(preB, inputs, rowBase, k + 192, cb);
    ee_compute(&lsA[1][0], embSw, kb0 + round * 2 + 2, halfSel, arow, ncol, acc);
  }

  // ---- write partial sums (D layout: VGPR rr -> M = rr + 8*(lane>=16)) ----
  float* outp = ps + (size_t)kc * M_ * H_;
  const int mAdd = halfSel * 8;
  #pragma unroll
  for (int s = 0; s < 4; ++s) {
    #pragma unroll
    for (int rr = 0; rr < 8; ++rr) {
      int row = m0 + s * 16 + rr + mAdd;
      outp[(size_t)row * H_ + ncol] = acc[s][rr];
    }
  }

  // ---- reduce per-row mask counts (4 threads per row) ----
  lsCnt[t] = (float)cntAcc;
  __syncthreads();
  if (q == 0) {
    float c = lsCnt[t] + lsCnt[t + 1] + lsCnt[t + 2] + lsCnt[t + 3];
    pc[(size_t)kc * M_ + mrow] = c;
  }
}

// ---------------------------------------------------------------------------
// Kernel 2: reduce chunk partials -> path_emb[m][h] (with cnt division).
// ---------------------------------------------------------------------------
__global__ __launch_bounds__(128)
void ee_reduce(const float* __restrict__ ps, const float* __restrict__ pc,
               float* __restrict__ pe /*[M_][H_]*/) {
  __shared__ float scnt;
  const int m = blockIdx.x;
  const int h = threadIdx.x;
  if (h == 0) {
    float c = 0.f;
    for (int ch = 0; ch < KC; ++ch) c += pc[(size_t)ch * M_ + m];
    scnt = c;
  }
  __syncthreads();
  float s = 0.f;
  for (int ch = 0; ch < KC; ++ch)
    s += ps[((size_t)ch * M_ + m) * H_ + h];
  float c = scnt;
  pe[(size_t)m * H_ + h] = (c > 0.f) ? s / fmaxf(c, 1.f) : 0.f;
}

// ---------------------------------------------------------------------------
// Kernel 3: mean over paths + LN -> GEMM+ReLU -> BN -> LN -> GEMM+ReLU -> BN.
// ---------------------------------------------------------------------------
__global__ __launch_bounds__(256)
void ee_mlp(const float* __restrict__ pe,
            const float* __restrict__ w1, const float* __restrict__ b1,
            const float* __restrict__ w2, const float* __restrict__ b2,
            const float* __restrict__ ln1g, const float* __restrict__ ln1b,
            const float* __restrict__ ln2g, const float* __restrict__ ln2b,
            const float* __restrict__ bn1g, const float* __restrict__ bn1b,
            const float* __restrict__ bn2g, const float* __restrict__ bn2b,
            float* __restrict__ out) {
  __shared__ float sX[B_ * H_];
  __shared__ float sY[B_ * H_];
  __shared__ float sMu[B_], sRs[B_];
  const int t = threadIdx.x;
  const float bnscale = rsqrtf(1.f + EPS_);

  #pragma unroll
  for (int j = 0; j < 16; ++j) {
    int idx = j * 256 + t;
    int b = idx >> 7, h = idx & (H_ - 1);
    float a = 0.f;
    for (int p = 0; p < P_; ++p)
      a += pe[(size_t)(b * P_ + p) * H_ + h];
    sX[idx] = a * (1.f / P_);
  }
  __syncthreads();

  if (t < B_) {
    float mu = 0.f;
    for (int h = 0; h < H_; ++h) mu += sX[t * H_ + h];
    mu *= (1.f / H_);
    float v = 0.f;
    for (int h = 0; h < H_; ++h) { float d = sX[t * H_ + h] - mu; v += d * d; }
    sMu[t] = mu; sRs[t] = rsqrtf(v * (1.f / H_) + EPS_);
  }
  __syncthreads();
  #pragma unroll
  for (int j = 0; j < 16; ++j) {
    int idx = j * 256 + t; int b = idx >> 7, h = idx & (H_ - 1);
    sX[idx] = ln1g[h] * (sX[idx] - sMu[b]) * sRs[b] + ln1b[h];
  }
  __syncthreads();

  #pragma unroll
  for (int j = 0; j < 16; ++j) {
    int idx = j * 256 + t; int b = idx >> 7, i = idx & (H_ - 1);
    float a = 0.f;
    for (int h = 0; h < H_; ++h) a += sX[b * H_ + h] * w1[i * H_ + h];
    a = fmaxf(a + b1[i], 0.f);
    sY[idx] = bn1g[i] * a * bnscale + bn1b[i];
  }
  __syncthreads();

  if (t < B_) {
    float mu = 0.f;
    for (int h = 0; h < H_; ++h) mu += sY[t * H_ + h];
    mu *= (1.f / H_);
    float v = 0.f;
    for (int h = 0; h < H_; ++h) { float d = sY[t * H_ + h] - mu; v += d * d; }
    sMu[t] = mu; sRs[t] = rsqrtf(v * (1.f / H_) + EPS_);
  }
  __syncthreads();
  #pragma unroll
  for (int j = 0; j < 16; ++j) {
    int idx = j * 256 + t; int b = idx >> 7, h = idx & (H_ - 1);
    sY[idx] = ln2g[h] * (sY[idx] - sMu[b]) * sRs[b] + ln2b[h];
  }
  __syncthreads();

  #pragma unroll
  for (int j = 0; j < 16; ++j) {
    int idx = j * 256 + t; int b = idx >> 7, i = idx & (H_ - 1);
    float a = 0.f;
    for (int h = 0; h < H_; ++h) a += sY[b * H_ + h] * w2[i * H_ + h];
    a = fmaxf(a + b2[i], 0.f);
    out[idx] = bn2g[i] * a * bnscale + bn2b[i];
  }
}

// ---------------------------------------------------------------------------
extern "C" void kernel_launch(void* const* d_in, const int* in_sizes, int n_in,
                              void* d_out, int out_size, void* d_ws, size_t ws_size,
                              hipStream_t stream) {
  const int*   inputs = (const int*)  d_in[0];
  const float* emb    = (const float*)d_in[1];
  const float* w1     = (const float*)d_in[2];
  const float* b1     = (const float*)d_in[3];
  const float* w2     = (const float*)d_in[4];
  const float* b2     = (const float*)d_in[5];
  const float* ln1g   = (const float*)d_in[6];
  const float* ln1b   = (const float*)d_in[7];
  const float* ln2g   = (const float*)d_in[8];
  const float* ln2b   = (const float*)d_in[9];
  const float* bn1g   = (const float*)d_in[10];
  const float* bn1b   = (const float*)d_in[11];
  const float* bn2g   = (const float*)d_in[12];
  const float* bn2b   = (const float*)d_in[13];

  float* ps = (float*)d_ws;                       // [KC][512][128]  (7 MB)
  float* pc = ps + (size_t)KC * M_ * H_;          // [KC][512]       (56 KB)
  float* pe = pc + (size_t)KC * M_;               // [512][128]      (256 KB)
  _Float16* embSw = (_Float16*)(pe + (size_t)M_ * H_);  // [KB32][2][128][16] (12.85 MB)

  ee_embswizzle<<<dim3(KB32), 256, 0, stream>>>(emb, embSw);
  ee_wmma_partial<<<dim3(M_ / 64, KC), 256, 0, stream>>>(inputs, embSw, ps, pc);
  ee_reduce<<<dim3(M_), 128, 0, stream>>>(ps, pc, pe);
  ee_mlp<<<1, 256, 0, stream>>>(pe, w1, b1, w2, b2,
                                ln1g, ln1b, ln2g, ln2b,
                                bn1g, bn1b, bn2g, bn2b, (float*)d_out);
}
